// GausskeyL2P_16037407883510
// MI455X (gfx1250) — matrix operand order repaired and checked by reading, not compiled, not run on hardware
//
#include <hip/hip_runtime.h>
#include <hip/hip_bf16.h>
#include <math.h>

// ---------------------------------------------------------------------------
// Types for CDNA5 WMMA (wave32, 16x16x32 bf16 -> f32)
// ---------------------------------------------------------------------------
typedef __bf16 bf16;
typedef __attribute__((ext_vector_type(8)))  __bf16 v8bf;
typedef __attribute__((ext_vector_type(16))) __bf16 v16bf;
typedef __attribute__((ext_vector_type(8)))  float  v8f;

// Model dims
#define Bsz   32
#define Dm    768
#define Hh    12
#define HD    64
#define MLPD  3072
#define SP    224          // padded sequence (multiple of 32), covers 197 and 222
#define MP    (Bsz*SP)     // 7168 = 56*128
#define NPAT  196
#define MPAT  (Bsz*NPAT)   // 6272 = 49*128
#define POOLN 10
#define SELN  5
#define PLENN 5
#define NCLS  100

__device__ __forceinline__ bf16 f2bf(float f) {
  unsigned u = __builtin_bit_cast(unsigned, f);
  unsigned r = (u + 0x7FFFu + ((u >> 16) & 1u)) >> 16;
  unsigned short s = (unsigned short)r;
  return __builtin_bit_cast(bf16, s);
}

__device__ __forceinline__ v8f vzero8() {
  v8f z = {0.f,0.f,0.f,0.f,0.f,0.f,0.f,0.f};
  return z;
}

// ---------------------------------------------------------------------------
// Weight convert + transpose: W[K][N] f32  ->  Wt[N][K] bf16
// ---------------------------------------------------------------------------
__global__ __launch_bounds__(256)
void convert_wt(const float* __restrict__ W, bf16* __restrict__ Wt, int K, int N) {
  size_t idx = (size_t)blockIdx.x * 256 + threadIdx.x;
  size_t tot = (size_t)K * N;
  if (idx >= tot) return;
  int n = (int)(idx % N);
  int k = (int)(idx / N);
  Wt[(size_t)n * K + k] = f2bf(W[(size_t)k * N + n]);
}

// ---------------------------------------------------------------------------
// Patchify: image [B,3,224,224] -> P bf16 [B*196, 768]  (col = c*256+py*16+px)
// ---------------------------------------------------------------------------
__global__ __launch_bounds__(256)
void patchify(const float* __restrict__ img, bf16* __restrict__ P) {
  size_t idx = (size_t)blockIdx.x * 256 + threadIdx.x;
  if (idx >= (size_t)MPAT * Dm) return;
  int m   = (int)(idx / Dm);
  int col = (int)(idx % Dm);
  int b = m / NPAT, pi = m % NPAT;
  int gy = pi / 14, gx = pi % 14;
  int c = col / 256, r2 = col % 256;
  int py = r2 / 16, px = r2 % 16;
  float v = img[(((size_t)b * 3 + c) * 224 + (gy * 16 + py)) * 224 + (gx * 16 + px)];
  P[idx] = f2bf(v);
}

// ---------------------------------------------------------------------------
// Generic bf16 WMMA GEMM: C[M,N] = A[M,K] @ Bt[N,K]^T (+bias)(+resid)(gelu)
// Block = 256 threads (8 waves). Tile 128x128, k-step 32.
// Wave w: rows 64*(w>>2), cols 32*(w&3); 4 M-frags x 2 N-frags.
// ---------------------------------------------------------------------------
__global__ __launch_bounds__(256)
void gemm_bf16_wmma(const bf16* __restrict__ A, const bf16* __restrict__ Bt,
                    const float* __restrict__ bias, const float* __restrict__ resid,
                    float* __restrict__ Cf, bf16* __restrict__ Cb,
                    int M, int N, int K, int gelu) {
  __shared__ bf16 As[128][32];
  __shared__ bf16 Bs[128][32];

  const int tid  = threadIdx.x;
  const int lane = tid & 31;
  const int w    = tid >> 5;
  const int wm   = (w >> 2) * 64;   // 0 or 64
  const int wn   = (w & 3) * 32;    // 0,32,64,96
  const int rowBlk = blockIdx.x * 128;
  const int colBlk = blockIdx.y * 128;

  v8f acc[4][2];
#pragma unroll
  for (int mf = 0; mf < 4; ++mf)
#pragma unroll
    for (int nf = 0; nf < 2; ++nf) acc[mf][nf] = vzero8();

  const int lr = tid >> 1;           // 0..127
  const int lc = (tid & 1) * 16;     // 0 or 16
  const int halfSel8  = (lane >= 16) ? 8 : 0;
  const int halfSel16 = (lane >= 16) ? 16 : 0;
  const int lrow = lane & 15;

  for (int k0 = 0; k0 < K; k0 += 32) {
    // Stage tiles (32B vector loads, fully coalesced; prefetch next tile)
    *(v16bf*)&As[lr][lc] = *(const v16bf*)&A[(size_t)(rowBlk + lr) * K + k0 + lc];
    *(v16bf*)&Bs[lr][lc] = *(const v16bf*)&Bt[(size_t)(colBlk + lr) * K + k0 + lc];
    if (k0 + 32 < K) {
      __builtin_prefetch(&A[(size_t)(rowBlk + lr) * K + k0 + 32 + lc], 0, 3);
      __builtin_prefetch(&Bt[(size_t)(colBlk + lr) * K + k0 + 32 + lc], 0, 3);
    }
    __syncthreads();

    // B fragments: halves i -> K = halfSel16 + i, lane%16 -> N
    v16bf bfrag[2];
#pragma unroll
    for (int nf = 0; nf < 2; ++nf)
      bfrag[nf] = *(const v16bf*)&Bs[wn + nf * 16 + lrow][halfSel16];

#pragma unroll
    for (int mf = 0; mf < 4; ++mf) {
      // A fragment: halves i -> K = halfSel16 + i
      v8bf alo = *(const v8bf*)&As[wm + mf * 16 + lrow][halfSel16];
      v8bf ahi = *(const v8bf*)&As[wm + mf * 16 + lrow][halfSel16 + 8];
      v16bf afrag;
#pragma unroll
      for (int i = 0; i < 8; ++i) { afrag[i] = alo[i]; afrag[8 + i] = ahi[i]; }
#pragma unroll
      for (int nf = 0; nf < 2; ++nf)
        acc[mf][nf] = __builtin_amdgcn_wmma_f32_16x16x32_bf16(
            false, afrag, false, bfrag[nf], (short)0, acc[mf][nf], false, false);
    }
    __syncthreads();
  }

  // Epilogue: lane -> N = lane%16, reg r -> M = r + (lane>=16?8:0)
  const int mofs = (lane >= 16) ? 8 : 0;
  const int ncol = lane & 15;
#pragma unroll
  for (int mf = 0; mf < 4; ++mf)
#pragma unroll
    for (int nf = 0; nf < 2; ++nf)
#pragma unroll
      for (int r = 0; r < 8; ++r) {
        int row = rowBlk + wm + mf * 16 + mofs + r;
        int col = colBlk + wn + nf * 16 + ncol;
        float v = acc[mf][nf][r];
        if (bias)  v += bias[col];
        if (resid) v += resid[(size_t)row * N + col];
        if (gelu)  v = 0.5f * v * (1.0f + erff(v * 0.70710678118654752f));
        if (Cf) Cf[(size_t)row * N + col] = v;
        if (Cb) Cb[(size_t)row * N + col] = f2bf(v);
      }
}

// ---------------------------------------------------------------------------
// Assemble X0 [B,224,768]: token0 = cls+pos0, 1..196 = Pout+pos, pads zero
// ---------------------------------------------------------------------------
__global__ __launch_bounds__(256)
void assemble_x0(const float* __restrict__ Pout, const float* __restrict__ cls,
                 const float* __restrict__ pos, float* __restrict__ X0) {
  size_t idx = (size_t)blockIdx.x * 256 + threadIdx.x;
  if (idx >= (size_t)MP * Dm) return;
  int m = (int)(idx / Dm), d = (int)(idx % Dm);
  int b = m / SP, s = m % SP;
  float v = 0.f;
  if (s == 0)        v = cls[d] + pos[d];
  else if (s <= 196) v = Pout[((size_t)(b * NPAT + s - 1)) * Dm + d] + pos[(size_t)s * Dm + d];
  X0[idx] = v;
}

// ---------------------------------------------------------------------------
// LayerNorm over D=768, output bf16; pad tokens (s>=S) forced to zero.
// One block per row, 256 threads (3 elements/thread).
// ---------------------------------------------------------------------------
__global__ __launch_bounds__(256)
void ln_bf16(const float* __restrict__ X, const float* __restrict__ w,
             const float* __restrict__ bb, bf16* __restrict__ out, int S) {
  __shared__ float red[256];
  int m = blockIdx.x, tid = threadIdx.x;
  int s = m % SP;
  const float* x = X + (size_t)m * Dm;
  float v0 = x[tid], v1 = x[tid + 256], v2 = x[tid + 512];
  red[tid] = v0 + v1 + v2; __syncthreads();
  for (int o = 128; o; o >>= 1) { if (tid < o) red[tid] += red[tid + o]; __syncthreads(); }
  float mean = red[0] * (1.0f / Dm); __syncthreads();
  float d0 = v0 - mean, d1 = v1 - mean, d2 = v2 - mean;
  red[tid] = d0 * d0 + d1 * d1 + d2 * d2; __syncthreads();
  for (int o = 128; o; o >>= 1) { if (tid < o) red[tid] += red[tid + o]; __syncthreads(); }
  float inv = rsqrtf(red[0] * (1.0f / Dm) + 1e-6f);
  bf16* o_ = out + (size_t)m * Dm;
  if (s >= S) {
    o_[tid] = f2bf(0.f); o_[tid + 256] = f2bf(0.f); o_[tid + 512] = f2bf(0.f);
  } else {
    o_[tid]       = f2bf(d0 * inv * w[tid]       + bb[tid]);
    o_[tid + 256] = f2bf(d1 * inv * w[tid + 256] + bb[tid + 256]);
    o_[tid + 512] = f2bf(d2 * inv * w[tid + 512] + bb[tid + 512]);
  }
}

// ---------------------------------------------------------------------------
// QKV scatter into pre-swizzled WMMA fragment buffers.
// QKV f32 [MP, 2304]; fragments stored so one lane loads 16 contiguous halves.
// ---------------------------------------------------------------------------
// Q as A-fragments: [bh][qt(14)][ks(2)][lane(32)][i(16)]
__global__ __launch_bounds__(512)
void scatter_q(const float* __restrict__ QKV, bf16* __restrict__ Qf, int S) {
  int tid = threadIdx.x;
  int i = tid & 15, lane = tid >> 4;
  int blk = blockIdx.x;                 // bh*28 + qt*2 + ks
  int ks = blk & 1, qt = (blk >> 1) % 14, bh = blk / 28;
  int b = bh / Hh, h = bh % Hh;
  int tok = qt * 16 + (lane & 15);
  int d = ks * 32 + ((lane >= 16) ? 16 : 0) + i;
  float v = (tok < S) ? QKV[((size_t)(b * SP + tok)) * 2304 + (h * HD + d)] : 0.f;
  Qf[(size_t)blk * 512 + tid] = f2bf(v);
}

// K^T as B-fragments (scores): [bh][kt(14)][ks(2)][lane][i]; K = d, N = token
__global__ __launch_bounds__(512)
void scatter_kt(const float* __restrict__ QKV, bf16* __restrict__ Ktf, int S) {
  int tid = threadIdx.x;
  int i = tid & 15, lane = tid >> 4;
  int blk = blockIdx.x;
  int ks = blk & 1, kt = (blk >> 1) % 14, bh = blk / 28;
  int b = bh / Hh, h = bh % Hh;
  int tok = kt * 16 + (lane & 15);
  int d = ks * 32 + ((lane >= 16) ? 16 : 0) + i;
  float v = (tok < S) ? QKV[((size_t)(b * SP + tok)) * 2304 + (Dm + h * HD + d)] : 0.f;
  Ktf[(size_t)blk * 512 + tid] = f2bf(v);
}

// V as B-fragments (P@V): [bh][sc(7)][nt(4)][lane][i]; K = token, N = d
__global__ __launch_bounds__(512)
void scatter_v(const float* __restrict__ QKV, bf16* __restrict__ Vf, int S) {
  int tid = threadIdx.x;
  int i = tid & 15, lane = tid >> 4;
  int blk = blockIdx.x;                 // bh*28 + sc*4 + nt
  int nt = blk & 3, sc = (blk >> 2) % 7, bh = blk / 28;
  int b = bh / Hh, h = bh % Hh;
  int d = nt * 16 + (lane & 15);
  int tok = sc * 32 + (i >> 3) * 16 + ((lane >= 16) ? 8 : 0) + (i & 7);
  float v = (tok < S) ? QKV[((size_t)(b * SP + tok)) * 2304 + (2 * Dm + h * HD + d)] : 0.f;
  Vf[(size_t)blk * 512 + tid] = f2bf(v);
}

// ---------------------------------------------------------------------------
// Attention: one wave per (q-tile of 16 rows, b, h).
// scores(WMMA) -> LDS f32 -> masked softmax -> P@V (WMMA) -> Obf bf16 [MP,768]
// ---------------------------------------------------------------------------
__global__ __launch_bounds__(32)
void attention_wmma(const bf16* __restrict__ Qf, const bf16* __restrict__ Ktf,
                    const bf16* __restrict__ Vf, bf16* __restrict__ Obf, int S) {
  __shared__ float sc[16][SP];
  const int qt = blockIdx.x;       // 0..13
  const int bh = blockIdx.y;       // 0..383
  const int b = bh / Hh, h = bh % Hh;
  const int lane = threadIdx.x;
  const int rofs = (lane >= 16) ? 8 : 0;
  const int ncol = lane & 15;

  // Q A-fragments (2 k-steps over HD=64)
  size_t qb = ((size_t)(bh * 14 + qt) * 2) * 512;
  v16bf qf0 = *(const v16bf*)&Qf[qb + (size_t)lane * 16];
  v16bf qf1 = *(const v16bf*)&Qf[qb + 512 + (size_t)lane * 16];

  // Scores: 14 tiles of 16 key-tokens
  for (int kt = 0; kt < 14; ++kt) {
    size_t kb = ((size_t)(bh * 14 + kt) * 2) * 512;
    v16bf k0 = *(const v16bf*)&Ktf[kb + (size_t)lane * 16];
    v16bf k1 = *(const v16bf*)&Ktf[kb + 512 + (size_t)lane * 16];
    v8f a = vzero8();
    a = __builtin_amdgcn_wmma_f32_16x16x32_bf16(false, qf0, false, k0, (short)0, a, false, false);
    a = __builtin_amdgcn_wmma_f32_16x16x32_bf16(false, qf1, false, k1, (short)0, a, false, false);
    int col = kt * 16 + ncol;
#pragma unroll
    for (int r = 0; r < 8; ++r) sc[rofs + r][col] = a[r];
  }
  __syncthreads();

  // Softmax (scale 1/sqrt(64)=0.125), lanes 0..15 own one row each
  if (lane < 16) {
    float mx = -1e30f;
    for (int c = 0; c < S; ++c) mx = fmaxf(mx, sc[lane][c] * 0.125f);
    float sum = 0.f;
    for (int c = 0; c < S; ++c) {
      float e = expf(sc[lane][c] * 0.125f - mx);
      sc[lane][c] = e; sum += e;
    }
    float inv = 1.0f / sum;
    for (int c = 0; c < S; ++c) sc[lane][c] *= inv;
    for (int c = S; c < SP; ++c) sc[lane][c] = 0.f;
  }
  __syncthreads();

  // O = P @ V : 7 chunks of 32 tokens, 4 d-tiles of 16
  v8f o0 = vzero8(), o1 = vzero8(), o2 = vzero8(), o3 = vzero8();
  for (int scx = 0; scx < 7; ++scx) {
    const float* sr = &sc[ncol][scx * 32 + rofs];  // A-frag: row=lane%16, K pattern
    v16bf af;
#pragma unroll
    for (int i = 0; i < 8; ++i) { af[i] = f2bf(sr[i]); af[8 + i] = f2bf(sr[16 + i]); }
    size_t vb = ((size_t)(bh * 7 + scx) * 4) * 512;
    v16bf vf0 = *(const v16bf*)&Vf[vb + 0 * 512 + (size_t)lane * 16];
    v16bf vf1 = *(const v16bf*)&Vf[vb + 1 * 512 + (size_t)lane * 16];
    v16bf vf2 = *(const v16bf*)&Vf[vb + 2 * 512 + (size_t)lane * 16];
    v16bf vf3 = *(const v16bf*)&Vf[vb + 3 * 512 + (size_t)lane * 16];
    o0 = __builtin_amdgcn_wmma_f32_16x16x32_bf16(false, af, false, vf0, (short)0, o0, false, false);
    o1 = __builtin_amdgcn_wmma_f32_16x16x32_bf16(false, af, false, vf1, (short)0, o1, false, false);
    o2 = __builtin_amdgcn_wmma_f32_16x16x32_bf16(false, af, false, vf2, (short)0, o2, false, false);
    o3 = __builtin_amdgcn_wmma_f32_16x16x32_bf16(false, af, false, vf3, (short)0, o3, false, false);
  }

#pragma unroll
  for (int r = 0; r < 8; ++r) {
    int q = qt * 16 + rofs + r;
    size_t base = ((size_t)(b * SP + q)) * Dm + h * HD;
    Obf[base + 0 * 16 + ncol] = f2bf(o0[r]);
    Obf[base + 1 * 16 + ncol] = f2bf(o1[r]);
    Obf[base + 2 * 16 + ncol] = f2bf(o2[r]);
    Obf[base + 3 * 16 + ncol] = f2bf(o3[r]);
  }
}

// ---------------------------------------------------------------------------
// CLS LayerNorm (f32 out): q[b][768] = LN(X[b*224])
// ---------------------------------------------------------------------------
__global__ __launch_bounds__(256)
void cls_ln(const float* __restrict__ X, const float* __restrict__ w,
            const float* __restrict__ bb, float* __restrict__ q) {
  __shared__ float red[256];
  int b = blockIdx.x, tid = threadIdx.x;
  const float* x = X + (size_t)(b * SP) * Dm;
  float v0 = x[tid], v1 = x[tid + 256], v2 = x[tid + 512];
  red[tid] = v0 + v1 + v2; __syncthreads();
  for (int o = 128; o; o >>= 1) { if (tid < o) red[tid] += red[tid + o]; __syncthreads(); }
  float mean = red[0] * (1.0f / Dm); __syncthreads();
  float d0 = v0 - mean, d1 = v1 - mean, d2 = v2 - mean;
  red[tid] = d0 * d0 + d1 * d1 + d2 * d2; __syncthreads();
  for (int o = 128; o; o >>= 1) { if (tid < o) red[tid] += red[tid + o]; __syncthreads(); }
  float inv = rsqrtf(red[0] * (1.0f / Dm) + 1e-6f);
  float* o_ = q + (size_t)b * Dm;
  o_[tid]       = d0 * inv * w[tid]       + bb[tid];
  o_[tid + 256] = d1 * inv * w[tid + 256] + bb[tid + 256];
  o_[tid + 512] = d2 * inv * w[tid + 512] + bb[tid + 512];
}

// 0.5*log(||var_p||_F) = 0.25*log(sum of squares)
__global__ __launch_bounds__(256)
void lognorm_k(const float* __restrict__ var, float* __restrict__ logn) {
  __shared__ float red[256];
  int p = blockIdx.x, tid = threadIdx.x;
  const float* v = var + (size_t)p * Dm * Dm;
  float s = 0.f;
  for (size_t i = tid; i < (size_t)Dm * Dm; i += 256) { float x = v[i]; s += x * x; }
  red[tid] = s; __syncthreads();
  for (int o = 128; o; o >>= 1) { if (tid < o) red[tid] += red[tid + o]; __syncthreads(); }
  if (tid == 0) logn[p] = 0.25f * logf(red[0]);
}

// quad[b][p] = diff^T invV_p diff
__global__ __launch_bounds__(256)
void quad_k(const float* __restrict__ q, const float* __restrict__ kp,
            const float* __restrict__ inv, float* __restrict__ quad) {
  __shared__ float diff[Dm];
  __shared__ float red[256];
  int p = blockIdx.x, b = blockIdx.y, tid = threadIdx.x;
  for (int i = tid; i < Dm; i += 256)
    diff[i] = q[(size_t)b * Dm + i] - kp[(size_t)p * Dm + i];
  __syncthreads();
  const float* M = inv + (size_t)p * Dm * Dm;
  float acc = 0.f;
  for (int e = tid; e < Dm; e += 256) {
    float t = 0.f;
    for (int d = 0; d < Dm; ++d) t += diff[d] * M[(size_t)d * Dm + e];
    acc += t * diff[e];
  }
  red[tid] = acc; __syncthreads();
  for (int o = 128; o; o >>= 1) { if (tid < o) red[tid] += red[tid + o]; __syncthreads(); }
  if (tid == 0) quad[b * POOLN + p] = red[0];
}

// score + top-5 smallest (ascending, ties -> lower index), one thread per batch
__global__ __launch_bounds__(32)
void score_topk(const float* __restrict__ quad, const float* __restrict__ logn,
                const float* __restrict__ freq, int* __restrict__ topk) {
  int b = threadIdx.x;
  if (b >= Bsz) return;
  float fm = -1e30f;
  for (int p = 0; p < POOLN; ++p) fm = fmaxf(fm, freq[p]);
  float nrm = 0.f;
  for (int p = 0; p < POOLN; ++p) { float wv = fm - freq[p]; nrm += wv * wv; }
  nrm = fmaxf(sqrtf(nrm), 1e-12f);
  float sc[POOLN];
  for (int p = 0; p < POOLN; ++p) {
    float wn = (fm - freq[p]) / nrm;
    sc[p] = -0.5f * quad[b * POOLN + p] + logn[p] + logf(fmaxf(wn, 1e-30f));
  }
  bool used[POOLN] = {};
  for (int j = 0; j < SELN; ++j) {
    int bi = 0; float bs = 1e38f;
    for (int p = 0; p < POOLN; ++p)
      if (!used[p] && sc[p] < bs) { bs = sc[p]; bi = p; }
    used[bi] = true;
    topk[b * SELN + j] = bi;
  }
}

// SX [B,224,768]: [cls, 25 prompts(+pos0), 196 patch tokens], pads zero
__global__ __launch_bounds__(256)
void assemble_sx(const float* __restrict__ X0, const float* __restrict__ prompts,
                 const float* __restrict__ pos, const int* __restrict__ topk,
                 float* __restrict__ SX) {
  size_t idx = (size_t)blockIdx.x * 256 + threadIdx.x;
  if (idx >= (size_t)MP * Dm) return;
  int m = (int)(idx / Dm), d = (int)(idx % Dm);
  int b = m / SP, s = m % SP;
  float v = 0.f;
  if (s == 0) {
    v = X0[((size_t)(b * SP)) * Dm + d];
  } else if (s <= SELN * PLENN) {
    int i = s - 1;
    int pidx = topk[b * SELN + i / PLENN];
    v = prompts[((size_t)pidx * PLENN + (i % PLENN)) * Dm + d] + pos[d];
  } else if (s < 222) {
    v = X0[((size_t)(b * SP + (s - 25))) * Dm + d];
  }
  SX[idx] = v;
}

// Final LN over tokens 1..25 then mean -> feat[b][768]
__global__ __launch_bounds__(256)
void final_ln_mean(const float* __restrict__ SX, const float* __restrict__ w,
                   const float* __restrict__ bb, float* __restrict__ feat) {
  __shared__ float red[256];
  int b = blockIdx.x, tid = threadIdx.x;
  float a0 = 0.f, a1 = 0.f, a2 = 0.f;
  for (int t = 1; t <= SELN * PLENN; ++t) {
    const float* x = SX + ((size_t)(b * SP + t)) * Dm;
    float v0 = x[tid], v1 = x[tid + 256], v2 = x[tid + 512];
    red[tid] = v0 + v1 + v2; __syncthreads();
    for (int o = 128; o; o >>= 1) { if (tid < o) red[tid] += red[tid + o]; __syncthreads(); }
    float mean = red[0] * (1.0f / Dm); __syncthreads();
    float d0 = v0 - mean, d1 = v1 - mean, d2 = v2 - mean;
    red[tid] = d0 * d0 + d1 * d1 + d2 * d2; __syncthreads();
    for (int o = 128; o; o >>= 1) { if (tid < o) red[tid] += red[tid + o]; __syncthreads(); }
    float inv = rsqrtf(red[0] * (1.0f / Dm) + 1e-6f); __syncthreads();
    a0 += d0 * inv * w[tid]       + bb[tid];
    a1 += d1 * inv * w[tid + 256] + bb[tid + 256];
    a2 += d2 * inv * w[tid + 512] + bb[tid + 512];
  }
  const float sc = 1.0f / (SELN * PLENN);
  feat[(size_t)b * Dm + tid]       = a0 * sc;
  feat[(size_t)b * Dm + tid + 256] = a1 * sc;
  feat[(size_t)b * Dm + tid + 512] = a2 * sc;
}

// Head: out[b][k] = feat[b] . head_w[:,k] + head_b[k]
__global__ __launch_bounds__(128)
void head_k(const float* __restrict__ feat, const float* __restrict__ hw,
            const float* __restrict__ hb, float* __restrict__ out) {
  int b = blockIdx.x, k = threadIdx.x;
  if (k >= NCLS) return;
  const float* f = feat + (size_t)b * Dm;
  float s = 0.f;
  for (int d = 0; d < Dm; ++d) s += f[d] * hw[(size_t)d * NCLS + k];
  out[(size_t)b * NCLS + k] = s + hb[k];
}

// ---------------------------------------------------------------------------
// Host orchestration
// ---------------------------------------------------------------------------
extern "C" void kernel_launch(void* const* d_in, const int* in_sizes, int n_in,
                              void* d_out, int out_size, void* d_ws, size_t ws_size,
                              hipStream_t stream) {
  const float* inputs   = (const float*)d_in[0];
  const float* patch_w  = (const float*)d_in[1];
  const float* patch_b  = (const float*)d_in[2];
  const float* cls_tok  = (const float*)d_in[3];
  const float* pos_emb  = (const float*)d_in[4];
  const float* ln1_w    = (const float*)d_in[5];
  const float* ln1_b    = (const float*)d_in[6];
  const float* qkv_w    = (const float*)d_in[7];
  const float* qkv_b    = (const float*)d_in[8];
  const float* proj_w   = (const float*)d_in[9];
  const float* proj_b   = (const float*)d_in[10];
  const float* ln2_w    = (const float*)d_in[11];
  const float* ln2_b    = (const float*)d_in[12];
  const float* fc1_w    = (const float*)d_in[13];
  const float* fc1_b    = (const float*)d_in[14];
  const float* fc2_w    = (const float*)d_in[15];
  const float* fc2_b    = (const float*)d_in[16];
  const float* norm_w   = (const float*)d_in[17];
  const float* norm_b   = (const float*)d_in[18];
  const float* head_w   = (const float*)d_in[19];
  const float* head_b   = (const float*)d_in[20];
  const float* key_pool = (const float*)d_in[21];
  const float* freqency = (const float*)d_in[22];
  const float* prompts  = (const float*)d_in[23];
  const float* variance = (const float*)d_in[24];
  const float* inv_var  = (const float*)d_in[25];

  char* ws = (char*)d_ws;
  size_t off = 0;
  auto alloc = [&](size_t bytes) -> void* {
    void* p = ws + off;
    off += (bytes + 255) & ~(size_t)255;
    return p;
  };

  bf16* patch_wt = (bf16*)alloc((size_t)Dm * Dm * 2);
  bf16* qkv_wt   = (bf16*)alloc((size_t)12 * 3 * Dm * Dm * 2);
  bf16* proj_wt  = (bf16*)alloc((size_t)12 * Dm * Dm * 2);
  bf16* fc1_wt   = (bf16*)alloc((size_t)12 * MLPD * Dm * 2);
  bf16* fc2_wt   = (bf16*)alloc((size_t)12 * Dm * MLPD * 2);
  bf16* Pbf      = (bf16*)alloc((size_t)MPAT * Dm * 2);
  float* Pout    = (float*)alloc((size_t)MPAT * Dm * 4);
  float* X0      = (float*)alloc((size_t)MP * Dm * 4);
  float* Xa      = (float*)alloc((size_t)MP * Dm * 4);
  float* Xb      = (float*)alloc((size_t)MP * Dm * 4);
  float* SX      = (float*)alloc((size_t)MP * Dm * 4);
  bf16* Hb       = (bf16*)alloc((size_t)MP * Dm * 2);
  float* QKV     = (float*)alloc((size_t)MP * 3 * Dm * 4);
  bf16* Qf       = (bf16*)alloc((size_t)Bsz * Hh * 14 * 2 * 512 * 2);
  bf16* Ktf      = (bf16*)alloc((size_t)Bsz * Hh * 14 * 2 * 512 * 2);
  bf16* Vf       = (bf16*)alloc((size_t)Bsz * Hh * 7 * 4 * 512 * 2);
  bf16* Obf      = (bf16*)alloc((size_t)MP * Dm * 2);
  bf16* Gb       = (bf16*)alloc((size_t)MP * MLPD * 2);
  float* qcls    = (float*)alloc((size_t)Bsz * Dm * 4);
  float* quadb   = (float*)alloc((size_t)Bsz * POOLN * 4);
  float* logn    = (float*)alloc((size_t)POOLN * 4);
  int*   topk    = (int*)alloc((size_t)Bsz * SELN * 4);
  float* featb   = (float*)alloc((size_t)Bsz * Dm * 4);
  (void)ws_size; (void)in_sizes; (void)n_in; (void)out_size;

  auto conv = [&](const float* W, bf16* Wt, int K, int N) {
    size_t tot = (size_t)K * N;
    convert_wt<<<(unsigned)((tot + 255) / 256), 256, 0, stream>>>(W, Wt, K, N);
  };

  // --- Weight conversion (f32 [K][N] -> bf16 [N][K]) ---
  conv(patch_w, patch_wt, Dm, Dm);
  for (int l = 0; l < 12; ++l) {
    conv(qkv_w  + (size_t)l * Dm * 3 * Dm, qkv_wt  + (size_t)l * 3 * Dm * Dm, Dm, 3 * Dm);
    conv(proj_w + (size_t)l * Dm * Dm,     proj_wt + (size_t)l * Dm * Dm,     Dm, Dm);
    conv(fc1_w  + (size_t)l * Dm * MLPD,   fc1_wt  + (size_t)l * MLPD * Dm,   Dm, MLPD);
    conv(fc2_w  + (size_t)l * MLPD * Dm,   fc2_wt  + (size_t)l * Dm * MLPD,   MLPD, Dm);
  }

  // --- Patch embedding ---
  patchify<<<(unsigned)(((size_t)MPAT * Dm + 255) / 256), 256, 0, stream>>>(inputs, Pbf);
  gemm_bf16_wmma<<<dim3(MPAT / 128, Dm / 128), 256, 0, stream>>>(
      Pbf, patch_wt, patch_b, nullptr, Pout, nullptr, MPAT, Dm, Dm, 0);
  assemble_x0<<<(unsigned)(((size_t)MP * Dm + 255) / 256), 256, 0, stream>>>(
      Pout, cls_tok, pos_emb, X0);
  hipMemcpyAsync(Xa, X0, (size_t)MP * Dm * 4, hipMemcpyDeviceToDevice, stream);

  auto run_pass = [&](float* cur, float* alt, int S) {
    for (int l = 0; l < 12; ++l) {
      ln_bf16<<<MP, 256, 0, stream>>>(cur, ln1_w + l * Dm, ln1_b + l * Dm, Hb, S);
      gemm_bf16_wmma<<<dim3(MP / 128, (3 * Dm) / 128), 256, 0, stream>>>(
          Hb, qkv_wt + (size_t)l * 3 * Dm * Dm, qkv_b + (size_t)l * 3 * Dm,
          nullptr, QKV, nullptr, MP, 3 * Dm, Dm, 0);
      scatter_q <<<Bsz * Hh * 28, 512, 0, stream>>>(QKV, Qf, S);
      scatter_kt<<<Bsz * Hh * 28, 512, 0, stream>>>(QKV, Ktf, S);
      scatter_v <<<Bsz * Hh * 28, 512, 0, stream>>>(QKV, Vf, S);
      attention_wmma<<<dim3(14, Bsz * Hh), 32, 0, stream>>>(Qf, Ktf, Vf, Obf, S);
      gemm_bf16_wmma<<<dim3(MP / 128, Dm / 128), 256, 0, stream>>>(
          Obf, proj_wt + (size_t)l * Dm * Dm, proj_b + (size_t)l * Dm,
          cur, alt, nullptr, MP, Dm, Dm, 0);
      ln_bf16<<<MP, 256, 0, stream>>>(alt, ln2_w + l * Dm, ln2_b + l * Dm, Hb, S);
      gemm_bf16_wmma<<<dim3(MP / 128, MLPD / 128), 256, 0, stream>>>(
          Hb, fc1_wt + (size_t)l * MLPD * Dm, fc1_b + (size_t)l * MLPD,
          nullptr, nullptr, Gb, MP, MLPD, Dm, 1);
      gemm_bf16_wmma<<<dim3(MP / 128, Dm / 128), 256, 0, stream>>>(
          Gb, fc2_wt + (size_t)l * Dm * MLPD, fc2_b + (size_t)l * Dm,
          alt, cur, nullptr, MP, Dm, MLPD, 0);
    }
  };

  // --- Pass 1 (S=197) ---
  run_pass(Xa, Xb, 197);

  // --- Mahalanobis routing + top-k ---
  cls_ln<<<Bsz, 256, 0, stream>>>(Xa, norm_w, norm_b, qcls);
  lognorm_k<<<POOLN, 256, 0, stream>>>(variance, logn);
  quad_k<<<dim3(POOLN, Bsz), 256, 0, stream>>>(qcls, key_pool, inv_var, quadb);
  score_topk<<<1, 32, 0, stream>>>(quadb, logn, freqency, topk);

  // --- Pass 2 with prompts (S=222) ---
  assemble_sx<<<(unsigned)(((size_t)MP * Dm + 255) / 256), 256, 0, stream>>>(
      X0, prompts, pos_emb, topk, SX);
  run_pass(SX, Xb, 222);

  // --- Final LN + prompt-token mean + head ---
  final_ln_mean<<<Bsz, 256, 0, stream>>>(SX, norm_w, norm_b, featb);
  head_k<<<Bsz, 128, 0, stream>>>(featb, head_w, head_b, (float*)d_out);
}
